// Encoder_52871047414535
// MI455X (gfx1250) — matrix operand order repaired and checked by reading, not compiled, not run on hardware
//
#include <hip/hip_runtime.h>
#include <math.h>

// ---------------------------------------------------------------------------
// Problem constants (from the reference): B=256 sets * 512 elems, DIM=128.
// val MLP (traced from gen_layer_sizes code): 128 -> 192 (LN+mish) -> 256
// key MLP: 512 -> 384 (LN+mish) -> 256, applied to one-hot rows => 512x256 table
// ---------------------------------------------------------------------------
#define NSETS   256
#define NPER    512
#define TOTAL   (NSETS * NPER)   // 131072
#define DIM     128
#define C1      192              // val hidden width
#define C2      256              // val output width
#define KH      384              // key hidden width
#define LN_EPS  1e-5f
#define ASTR    200              // LDS row stride (192 + 8 pad, halves)

typedef _Float16 v16h __attribute__((ext_vector_type(16)));
typedef float    v8f  __attribute__((ext_vector_type(8)));

// monotonic float<->uint encoding for atomicMax over signed floats
__device__ __forceinline__ unsigned fenc(float f) {
    unsigned u = __float_as_uint(f);
    return (u & 0x80000000u) ? ~u : (u | 0x80000000u);
}
__device__ __forceinline__ float fdec(unsigned u) {
    u = (u & 0x80000000u) ? (u & 0x7fffffffu) : ~u;
    return __uint_as_float(u);
}

__device__ __forceinline__ float mishf(float v) {
    float sp = (v > 20.f) ? v : log1pf(expf(v));   // softplus, overflow-safe
    return v * tanhf(sp);
}

// ---------------------------------------------------------------------------
// Kernel 1: convert val-MLP weights fp32->fp16, pre-swizzled into the CDNA5
// WMMA B fragments (16x16x32 f16), in this file's contiguous k order (matching A):  lane l, slot s  ->  K = 16*(l>>4)+s,
// N = l&15.  Fragment f = kt*(COUT/16)+nt; each lane's 16 halves contiguous.
// Also zero-inits the float-max accumulator.
// ---------------------------------------------------------------------------
__global__ void prep_kernel(const float* __restrict__ W0, const float* __restrict__ W1,
                            unsigned* maxbits, _Float16* __restrict__ W0f,
                            _Float16* __restrict__ W1f) {
    int idx = blockIdx.x * blockDim.x + threadIdx.x;
    if (idx == 0) *maxbits = 0u;
    const int N0 = DIM * C1;   // 24576
    const int N1 = C1 * C2;    // 49152
    if (idx < N0) {
        int s = idx & 15, lane = (idx >> 4) & 31, f = idx >> 9;
        int kt = f / (C1 / 16), nt = f % (C1 / 16);
        int K = kt * 32 + (lane >> 4) * 16 + s;
        int col = nt * 16 + (lane & 15);
        W0f[idx] = (_Float16)W0[K * C1 + col];
    } else if (idx < N0 + N1) {
        int j = idx - N0;
        int s = j & 15, lane = (j >> 4) & 31, f = j >> 9;
        int kt = f / (C2 / 16), nt = f % (C2 / 16);
        int K = kt * 32 + (lane >> 4) * 16 + s;
        int col = nt * 16 + (lane & 15);
        W1f[j] = (_Float16)W1[K * C2 + col];
    }
}

// ---------------------------------------------------------------------------
// Kernel 2: mag[e] = x[e,:]@rank_W + rank_b  (one wave per element) and
// global max via atomicMax on the monotonic encoding.
// ---------------------------------------------------------------------------
__global__ void mag_kernel(const float* __restrict__ x, const float* __restrict__ rankW,
                           const float* __restrict__ rankb, float* __restrict__ mag,
                           unsigned* maxbits) {
    int wid = threadIdx.x >> 5, lane = threadIdx.x & 31;
    int e = blockIdx.x * 8 + wid;
    float4 a = ((const float4*)(x + (size_t)e * DIM))[lane];
    float4 w = ((const float4*)rankW)[lane];
    float d = a.x * w.x + a.y * w.y + a.z * w.z + a.w * w.w;
    for (int off = 16; off; off >>= 1) d += __shfl_xor(d, off);
    if (lane == 0) {
        float v = d + rankb[0];
        mag[e] = v;
        atomicMax(maxbits, fenc(v));
    }
}

// ---------------------------------------------------------------------------
// Kernel 3: per-set stable rank on new_mag = mag + set*M (O(512^2) in LDS);
// scatter into per-set sorted arrays (values + original indices).
// ---------------------------------------------------------------------------
__global__ void rank_kernel(const float* __restrict__ mag, const unsigned* __restrict__ maxbits,
                            float* __restrict__ sortednew, int* __restrict__ sortedorig) {
    __shared__ float nm[NPER];
    int b = blockIdx.x, t = threadIdx.x;
    float M = fdec(*maxbits) + 1e-4f;
    int bo = b * NPER;
    for (int i = t; i < NPER; i += 256) nm[i] = mag[bo + i] + (float)b * M;
    __syncthreads();
    for (int i = t; i < NPER; i += 256) {
        float v = nm[i];
        int cnt = 0;
        for (int j = 0; j < NPER; j++) {
            float w = nm[j];
            cnt += (w < v) || (w == v && j < i);
        }
        sortednew[bo + cnt] = v;
        sortedorig[bo + cnt] = bo + i;
    }
}

// ---------------------------------------------------------------------------
// Kernel 4: global sorted position of each element (sets CAN interleave in the
// reference's global argsort).  For slot (b,r): g = r + sum over other sets of
// (#elems < v)  [+ #elems == v when b2 < b, matching stable-argsort tiebreak].
// Endpoint tests skip non-overlapping sets; only neighbors binary-search.
// Then invperm[g] = original element index.
// ---------------------------------------------------------------------------
__global__ void grank_kernel(const float* __restrict__ sortednew,
                             const int* __restrict__ sortedorig,
                             int* __restrict__ invperm) {
    int s = blockIdx.x * blockDim.x + threadIdx.x;
    int b = s >> 9, r = s & (NPER - 1);
    float v = sortednew[s];
    int g = r;
    for (int b2 = 0; b2 < NSETS; b2++) {
        if (b2 == b) continue;
        int base = b2 << 9;
        bool le = (b2 < b);
        float hi = sortednew[base + NPER - 1];
        if (le ? (hi <= v) : (hi < v)) { g += NPER; continue; }
        float lo = sortednew[base];
        if (le ? (lo > v) : (lo >= v)) continue;
        int L = 0, H = NPER;
        while (L < H) {
            int mid = (L + H) >> 1;
            float w = sortednew[base + mid];
            if (le ? (w <= v) : (w < v)) L = mid + 1; else H = mid;
        }
        g += L;
    }
    invperm[g] = sortedorig[s];
}

// ---------------------------------------------------------------------------
// Kernel 5: key table K[k,:] = key_mlp(one_hot(k)).  First layer on a one-hot
// is just W0[k,:] + b0; then LN(384)+mish; then 384x256 matmul.  Tiny: scalar.
// ---------------------------------------------------------------------------
__global__ void keytab_kernel(const float* __restrict__ W0, const float* __restrict__ b0,
                              const float* __restrict__ g0, const float* __restrict__ be0,
                              const float* __restrict__ W1, const float* __restrict__ b1,
                              float* __restrict__ Ktab) {
    __shared__ float h1[KH];
    __shared__ float ssum, ssq;
    int k = blockIdx.x, t = threadIdx.x;
    if (t == 0) { ssum = 0.f; ssq = 0.f; }
    for (int j = t; j < KH; j += 256) h1[j] = W0[k * KH + j] + b0[j];
    __syncthreads();
    float s = 0.f, q = 0.f;
    for (int j = t; j < KH; j += 256) { float v = h1[j]; s += v; q += v * v; }
    for (int off = 16; off; off >>= 1) { s += __shfl_xor(s, off); q += __shfl_xor(q, off); }
    if ((t & 31) == 0) { atomicAdd(&ssum, s); atomicAdd(&ssq, q); }
    __syncthreads();
    float mu = ssum * (1.f / KH);
    float var = ssq * (1.f / KH) - mu * mu;
    float rstd = rsqrtf(var + LN_EPS);
    for (int j = t; j < KH; j += 256) {
        float v = (h1[j] - mu) * rstd * g0[j] + be0[j];
        h1[j] = mishf(v);
    }
    __syncthreads();
    float acc = b1[t];
    for (int i = 0; i < KH; i++) acc += h1[i] * W1[i * C2 + t];
    Ktab[k * C2 + t] = acc;
}

// ---------------------------------------------------------------------------
// Kernel 6: out[b,c] = n[b]*card_W[c] + card_b[c]  (init; main kernel adds z).
// ---------------------------------------------------------------------------
__global__ void initout_kernel(const int* __restrict__ n, const float* __restrict__ cW,
                               const float* __restrict__ cb, float* __restrict__ out) {
    int b = blockIdx.x, c = threadIdx.x;
    out[b * C2 + c] = (float)n[b] * cW[c] + cb[c];
}

// ---------------------------------------------------------------------------
// Kernel 7 (main, WMMA): 128 threads = 4 waves; wave owns 16 sorted rows.
//   stage gathered x rows (fp16) -> L0 wmma (4x12) -> in-register LN+mish
//   -> L1 wmma (6x16) -> * K[key] -> per-wave column reduction -> atomicAdd.
// No block barriers: each wave uses only its own LDS stripe (DS in-order/wave).
// ---------------------------------------------------------------------------
__global__ void __launch_bounds__(128) main_kernel(
    const float* __restrict__ x, const int* __restrict__ invperm,
    const v16h* __restrict__ W0f, const v16h* __restrict__ W1f,
    const float* __restrict__ b0, const float* __restrict__ g0,
    const float* __restrict__ be0, const float* __restrict__ b1,
    const float* __restrict__ Ktab, float* __restrict__ out) {
    __shared__ _Float16 act[4 * 16 * ASTR];
    const int wave = threadIdx.x >> 5, lane = threadIdx.x & 31;
    const int h = lane >> 4, lm = lane & 15;
    const int p0 = blockIdx.x * 64 + wave * 16;
    _Float16* arow = act + wave * 16 * ASTR;

    // prefetch this block's K-table rows (hot in L2)
    for (int i = threadIdx.x; i < 64; i += 128)
        __builtin_prefetch(Ktab + (size_t)((blockIdx.x & 7) * 64 + i) * C2, 0, 1);

    // stage 16 gathered x rows, fp32 -> fp16 (one full row per iteration, coalesced)
    for (int it = 0; it < 16; ++it) {
        int orig = invperm[p0 + it];
        float4 v = ((const float4*)(x + (size_t)orig * DIM))[lane];
        _Float16* dst = arow + it * ASTR + lane * 4;
        dst[0] = (_Float16)v.x; dst[1] = (_Float16)v.y;
        dst[2] = (_Float16)v.z; dst[3] = (_Float16)v.w;
    }

    const v8f vz = {0.f, 0.f, 0.f, 0.f, 0.f, 0.f, 0.f, 0.f};

    // ---- layer 0: 128 -> 192 -------------------------------------------------
    v8f acc0[12];
#pragma unroll
    for (int i = 0; i < 12; ++i) acc0[i] = vz;
#pragma unroll
    for (int kt = 0; kt < 4; ++kt) {
        v16h A;
#pragma unroll
        for (int r = 0; r < 8; ++r) {             // contiguous A layout: K = 16*h + s
            int kk = 2 * r + 16 * h;
            int k = kt * 32 + kk;
            A[2 * r]     = arow[lm * ASTR + k];
            A[2 * r + 1] = arow[lm * ASTR + k + 1];
        }
#pragma unroll
        for (int nt = 0; nt < 12; ++nt) {
            v16h Bf = W0f[(kt * 12 + nt) * 32 + lane];   // pre-swizzled fragment
            acc0[nt] = __builtin_amdgcn_wmma_f32_16x16x32_f16(
                false, A, false, Bf, (short)0, acc0[nt], false, false);
        }
    }

    // bias + LayerNorm(192) + mish, back to LDS as fp16
    float bb[12], gg[12], bt[12];
#pragma unroll
    for (int nt = 0; nt < 12; ++nt) {
        int col = nt * 16 + lm;
        bb[nt] = b0[col]; gg[nt] = g0[col]; bt[nt] = be0[col];
    }
#pragma unroll
    for (int r = 0; r < 8; ++r) {                  // row M = r + 8*h
        float s = 0.f, q = 0.f;
#pragma unroll
        for (int nt = 0; nt < 12; ++nt) {
            float v = acc0[nt][r] + bb[nt];
            acc0[nt][r] = v; s += v; q += v * v;
        }
        s += __shfl_xor(s, 1); q += __shfl_xor(q, 1);
        s += __shfl_xor(s, 2); q += __shfl_xor(q, 2);
        s += __shfl_xor(s, 4); q += __shfl_xor(q, 4);
        s += __shfl_xor(s, 8); q += __shfl_xor(q, 8);   // reduce inside half-wave
        float mu = s * (1.0f / C1);
        float var = q * (1.0f / C1) - mu * mu;
        float rstd = rsqrtf(var + LN_EPS);
        int row = r + 8 * h;
#pragma unroll
        for (int nt = 0; nt < 12; ++nt) {
            int col = nt * 16 + lm;
            float v = (acc0[nt][r] - mu) * rstd * gg[nt] + bt[nt];
            arow[row * ASTR + col] = (_Float16)mishf(v);
        }
    }

    // ---- layer 1: 192 -> 256 -------------------------------------------------
    v8f acc1[16];
#pragma unroll
    for (int i = 0; i < 16; ++i) acc1[i] = vz;
#pragma unroll
    for (int kt = 0; kt < 6; ++kt) {
        v16h A;
#pragma unroll
        for (int r = 0; r < 8; ++r) {
            int kk = 2 * r + 16 * h;
            int k = kt * 32 + kk;
            A[2 * r]     = arow[lm * ASTR + k];
            A[2 * r + 1] = arow[lm * ASTR + k + 1];
        }
#pragma unroll
        for (int nt = 0; nt < 16; ++nt) {
            v16h Bf = W1f[(kt * 16 + nt) * 32 + lane];
            acc1[nt] = __builtin_amdgcn_wmma_f32_16x16x32_f16(
                false, A, false, Bf, (short)0, acc1[nt], false, false);
        }
    }

    // ---- epilogue: (val+b1) * K[key], reduce 16 rows, one atomic per column --
    const int keybase = (blockIdx.x & 7) * 64 + wave * 16;  // sorted pos mod 512
    const int set = blockIdx.x >> 3;                        // sorted pos / 512
#pragma unroll
    for (int nt = 0; nt < 16; ++nt) {
        int col = nt * 16 + lm;
        float bv = b1[col];
        float p = 0.f;
#pragma unroll
        for (int r = 0; r < 8; ++r) {
            int row = r + 8 * h;
            float kv = Ktab[(size_t)(keybase + row) * C2 + col];
            p += (acc1[nt][r] + bv) * kv;
        }
        p += __shfl_xor(p, 16);                  // combine the two half-waves
        if (h == 0) atomicAdd(out + (size_t)set * C2 + col, p);
    }
}

// ---------------------------------------------------------------------------
// Launcher.  Input order = setup_inputs() dict insertion order, leaves
// flattened depth-first (None g/beta of final layers dropped):
//  0:x 1:n | val: 2:W0 3:b0 4:g0 5:beta0 6:W1 7:b1 | key: 8:W0 9:b0 10:g0
//  11:beta0 12:W1 13:b1 | 14:rank_W 15:rank_b 16:card_W 17:card_b
// Workspace use: ~2.8 MB.
// ---------------------------------------------------------------------------
extern "C" void kernel_launch(void* const* d_in, const int* in_sizes, int n_in,
                              void* d_out, int out_size, void* d_ws, size_t ws_size,
                              hipStream_t stream) {
    (void)in_sizes; (void)n_in; (void)out_size; (void)ws_size;
    const float* x    = (const float*)d_in[0];
    const int*   n    = (const int*)d_in[1];
    const float* vW0  = (const float*)d_in[2];
    const float* vb0  = (const float*)d_in[3];
    const float* vg0  = (const float*)d_in[4];
    const float* vbe0 = (const float*)d_in[5];
    const float* vW1  = (const float*)d_in[6];
    const float* vb1  = (const float*)d_in[7];
    const float* kW0  = (const float*)d_in[8];
    const float* kb0  = (const float*)d_in[9];
    const float* kg0  = (const float*)d_in[10];
    const float* kbe0 = (const float*)d_in[11];
    const float* kW1  = (const float*)d_in[12];
    const float* kb1  = (const float*)d_in[13];
    const float* rankW = (const float*)d_in[14];
    const float* rankb = (const float*)d_in[15];
    const float* cardW = (const float*)d_in[16];
    const float* cardb = (const float*)d_in[17];
    float* out = (float*)d_out;

    char* ws = (char*)d_ws;
    size_t off = 0;
    auto carve = [&](size_t bytes) -> void* {
        size_t start = (off + 255) & ~(size_t)255;
        off = start + bytes;
        return (void*)(ws + start);
    };
    _Float16* W0f      = (_Float16*)carve((size_t)DIM * C1 * 2);
    _Float16* W1f      = (_Float16*)carve((size_t)C1 * C2 * 2);
    unsigned* maxbits  = (unsigned*)carve(4);
    float*    mag      = (float*)carve((size_t)TOTAL * 4);
    float*    sortednew= (float*)carve((size_t)TOTAL * 4);
    int*      sortedorig=(int*)  carve((size_t)TOTAL * 4);
    int*      invperm  = (int*)  carve((size_t)TOTAL * 4);
    float*    Ktab     = (float*)carve((size_t)NPER * C2 * 4);

    prep_kernel<<<(DIM * C1 + C1 * C2 + 255) / 256, 256, 0, stream>>>(vW0, vW1, maxbits, W0f, W1f);
    mag_kernel<<<TOTAL / 8, 256, 0, stream>>>(x, rankW, rankb, mag, maxbits);
    rank_kernel<<<NSETS, 256, 0, stream>>>(mag, maxbits, sortednew, sortedorig);
    grank_kernel<<<TOTAL / 256, 256, 0, stream>>>(sortednew, sortedorig, invperm);
    keytab_kernel<<<NPER, 256, 0, stream>>>(kW0, kb0, kg0, kbe0, kW1, kb1, Ktab);
    initout_kernel<<<NSETS, 256, 0, stream>>>(n, cardW, cardb, out);
    main_kernel<<<TOTAL / 64, 128, 0, stream>>>(x, invperm, (const v16h*)W0f, (const v16h*)W1f,
                                                vb0, vg0, vbe0, vb1, Ktab, out);
}
